// LrFeatureUpScaler_48000554500590
// MI455X (gfx1250) — matrix operand order repaired and checked by reading, not compiled, not run on hardware
//
#include <hip/hip_runtime.h>
#include <math.h>

// ---------------------------------------------------------------------------
// Problem constants (from the reference)
// ---------------------------------------------------------------------------
constexpr int kN  = 1024;   // nodes (also K of layer-1 input, and attention N)
constexpr int kCd = 512;    // per-head channel dim (both layers)

typedef __attribute__((ext_vector_type(16))) __bf16 v16bf;
typedef __attribute__((ext_vector_type(8)))  float  v8f;
typedef __bf16 bf16;

// fp32 -> bf16 round-to-nearest-even via bit trick (always compiles)
static __device__ __forceinline__ bf16 to_bf16(float f) {
  union { float f; unsigned u; } a; a.f = f;
  unsigned r = a.u + 0x7FFFu + ((a.u >> 16) & 1u);
  union { unsigned short s; bf16 b; } o; o.s = (unsigned short)(r >> 16);
  return o.b;
}

static __device__ __forceinline__ float blk_sum(float v, float* sm) {
  int t = threadIdx.x;
  sm[t] = v; __syncthreads();
  for (int s = blockDim.x >> 1; s > 0; s >>= 1) {
    if (t < s) sm[t] += sm[t + s];
    __syncthreads();
  }
  float r = sm[0]; __syncthreads();
  return r;
}

static __device__ __forceinline__ float blk_max(float v, float* sm) {
  int t = threadIdx.x;
  sm[t] = v; __syncthreads();
  for (int s = blockDim.x >> 1; s > 0; s >>= 1) {
    if (t < s) sm[t] = fmaxf(sm[t], sm[t + s]);
    __syncthreads();
  }
  float r = sm[0]; __syncthreads();
  return r;
}

// ---------------------------------------------------------------------------
// Pack fp32 row-major [M,K] into bf16 WMMA A-fragments (16x32 per fragment).
// Fragment f = mt*kTiles + kt, laid out [lane(32)][elem(16)] contiguous.
// ISA layout (16-bit A 16x32): lane<16  row=lane,    K = e + (e<8?0:8)
//                              lane>=16 row=lane-16, K = e + (e<8?8:16)
// ---------------------------------------------------------------------------
__global__ __launch_bounds__(256)
void pack_a_bf16(const float* __restrict__ src, int ld, int mTiles, int kTiles,
                 bf16* __restrict__ dst) {
  int idx = blockIdx.x * 256 + threadIdx.x;
  int total = mTiles * kTiles * 512;
  if (idx >= total) return;
  int f = idx >> 9, r = idx & 511;
  int lane = r >> 4, e = r & 15;
  int mt = f / kTiles, kt = f - mt * kTiles;
  int row = mt * 16 + (lane & 15);
  int k = kt * 32 + e + ((e < 8) ? 0 : 8) + ((lane < 16) ? 0 : 8);
  dst[idx] = to_bf16(src[(size_t)row * ld + k]);
}

// ---------------------------------------------------------------------------
// Pack fp32 into bf16 WMMA B-fragments (32x16 per fragment), f = nt*kTiles+kt.
// ISA layout (16-bit B 32x16): lane<16  col=lane,    K = e + (e<8?0:8)
//                              lane>=16 col=lane-16, K = e + (e<8?8:16)
// trans==0: element (k,n) = src[k*ld + n]   (weights, V)
// trans==1: element (k,n) = src[n*ld + k]   (K-matrix for Q@K^T)
// ---------------------------------------------------------------------------
__global__ __launch_bounds__(256)
void pack_b_bf16(const float* __restrict__ src, int ld, int kTiles, int nTiles,
                 int trans, bf16* __restrict__ dst) {
  int idx = blockIdx.x * 256 + threadIdx.x;
  int total = nTiles * kTiles * 512;
  if (idx >= total) return;
  int f = idx >> 9, r = idx & 511;
  int lane = r >> 4, e = r & 15;
  int nt = f / kTiles, kt = f - nt * kTiles;
  int n = nt * 16 + (lane & 15);
  int k = kt * 32 + e + ((e < 8) ? 0 : 8) + ((lane < 16) ? 0 : 8);
  float v = trans ? src[(size_t)n * ld + k] : src[(size_t)k * ld + n];
  dst[idx] = to_bf16(v);
}

// ---------------------------------------------------------------------------
// One wave computes 2 M-tiles x 4 N-tiles (32x64 of C); fp32 accumulate,
// optional fused bias.  k-loop unrolled by 2 with ping-pong fragment buffers:
// loads for the next k-step issue before the WMMAs of the current one, and
// the buffer swap is structural (no register copies, no hazard NOPs).
// kTiles must be even (true for all call sites: 16/32/64).
// ---------------------------------------------------------------------------
__global__ __launch_bounds__(32)
void gemm_bf16_wmma(const bf16* __restrict__ Ap, const bf16* __restrict__ Bp,
                    const float* __restrict__ bias, float* __restrict__ C,
                    int ldc, int kTiles) {
  int lane = threadIdx.x;
  int mt0 = blockIdx.x * 2;
  int ntBase = blockIdx.y * 4;

  v8f acc[8] = {};

  const bf16* aPtr[2];
  const bf16* bPtr[4];
#pragma unroll
  for (int i = 0; i < 2; ++i)
    aPtr[i] = Ap + (size_t)(mt0 + i) * kTiles * 512 + lane * 16;
#pragma unroll
  for (int j = 0; j < 4; ++j)
    bPtr[j] = Bp + (size_t)(ntBase + j) * kTiles * 512 + lane * 16;

  v16bf a0[2], b0[4], a1[2], b1[4];

  auto loadFrag = [&](v16bf (&aF)[2], v16bf (&bF)[4], int kt) {
#pragma unroll
    for (int i = 0; i < 2; ++i)
      aF[i] = *(const v16bf*)(aPtr[i] + (size_t)kt * 512);
#pragma unroll
    for (int j = 0; j < 4; ++j)
      bF[j] = *(const v16bf*)(bPtr[j] + (size_t)kt * 512);
  };
  auto mma8 = [&](const v16bf (&aF)[2], const v16bf (&bF)[4]) {
#pragma unroll
    for (int i = 0; i < 2; ++i)
#pragma unroll
      for (int j = 0; j < 4; ++j)
        acc[i * 4 + j] = __builtin_amdgcn_wmma_f32_16x16x32_bf16(
            false, aF[i], false, bF[j], (short)0, acc[i * 4 + j], false, false);
  };

  loadFrag(a0, b0, 0);

  int kt = 0;
  for (; kt + 2 < kTiles; kt += 2) {
    // stream prefetch for the B fragments a few k-steps ahead
#pragma unroll
    for (int j = 0; j < 4; ++j)
      __builtin_prefetch(bPtr[j] + (size_t)(kt + 4) * 512, 0, 1);
    loadFrag(a1, b1, kt + 1);
    mma8(a0, b0);
#pragma unroll
    for (int j = 0; j < 4; ++j)
      __builtin_prefetch(bPtr[j] + (size_t)(kt + 5) * 512, 0, 1);
    loadFrag(a0, b0, kt + 2);
    mma8(a1, b1);
  }
  // tail (kTiles even): kt == kTiles-2 here
  loadFrag(a1, b1, kt + 1);
  mma8(a0, b0);
  mma8(a1, b1);

  // f32 C/D layout: VGPR r, lane l -> row = r + (l<16?0:8), col = l&15
  int col = ntBase * 16 + (lane & 15);
#pragma unroll
  for (int m = 0; m < 2; ++m) {
    int row0 = (mt0 + m) * 16 + ((lane < 16) ? 0 : 8);
#pragma unroll
    for (int j = 0; j < 4; ++j) {
      float bb = bias ? bias[col + j * 16] : 0.0f;
#pragma unroll
      for (int i = 0; i < 8; ++i)
        C[(size_t)(row0 + i) * ldc + col + j * 16] = acc[m * 4 + j][i] + bb;
    }
  }
}

// qe[c,h] = dot(q[c, h*Cd : (h+1)*Cd], We[h,:])
__global__ __launch_bounds__(128)
void qe_dot(const float* __restrict__ q, const float* __restrict__ ew,
            float* __restrict__ qe, int H, int ldq) {
  __shared__ float sm[128];
  int c = blockIdx.x / H, h = blockIdx.x % H;
  const float* qr = q + (size_t)c * ldq + h * kCd;
  const float* er = ew + (size_t)h * kCd;
  float s = 0.f;
  for (int d = threadIdx.x; d < kCd; d += 128) s += qr[d] * er[d];
  s = blk_sum(s, sm);
  if (threadIdx.x == 0) qe[(size_t)c * H + h] = s;
}

// logits = (S + x[r,c]*qe[c,h]) * scale; softmax over r (in-place into S);
// beta[c,h] = sum_r P[c,r] * x[r,c].  One 256-thread block per (h,c) row.
__global__ __launch_bounds__(256)
void attn_softmax(float* __restrict__ S, const float* __restrict__ x,
                  const float* __restrict__ qe, float* __restrict__ beta,
                  int H, float scale) {
  __shared__ float sm[256];
  int h = blockIdx.x / kN, c = blockIdx.x % kN;
  float* row = S + ((size_t)h * kN + c) * kN;
  float qv = qe[(size_t)c * H + h];
  int tid = threadIdx.x;
  float t[4], l[4];
  float m = -3.0e38f;
#pragma unroll
  for (int i = 0; i < 4; ++i) {
    int r = tid + 256 * i;
    t[i] = x[(size_t)r * kN + c];
    l[i] = (row[r] + t[i] * qv) * scale;
    m = fmaxf(m, l[i]);
  }
  m = blk_max(m, sm);
  float s = 0.f;
#pragma unroll
  for (int i = 0; i < 4; ++i) { l[i] = __expf(l[i] - m); s += l[i]; }
  s = blk_sum(s, sm);
  float inv = 1.0f / s, bp = 0.f;
#pragma unroll
  for (int i = 0; i < 4; ++i) {
    float p = l[i] * inv;
    row[tid + 256 * i] = p;
    bp += p * t[i];
  }
  bp = blk_sum(bp, sm);
  if (tid == 0) beta[(size_t)c * H + h] = bp;
}

// out = attn + beta[c,h]*We + skip
__global__ __launch_bounds__(256)
void attn_epilogue(const float* __restrict__ attnv, const float* __restrict__ beta,
                   const float* __restrict__ ew, const float* __restrict__ skip,
                   float* __restrict__ out, int H, int Cd, int HR, int total) {
  int idx = blockIdx.x * 256 + threadIdx.x;
  if (idx >= total) return;
  int c = idx / HR, j = idx - c * HR;
  out[idx] = attnv[idx] + beta[(size_t)c * H + (j / Cd)] * ew[j] + skip[idx];
}

// GraphNorm over 1024 rows, one block per feature column.
__global__ __launch_bounds__(256)
void graph_norm(const float* __restrict__ in, float* __restrict__ out,
                const float* __restrict__ gamma, const float* __restrict__ bet,
                const float* __restrict__ ms, int HR) {
  __shared__ float sm[256];
  int j = blockIdx.x, tid = threadIdx.x;
  float v[4];
  float s = 0.f;
#pragma unroll
  for (int i = 0; i < 4; ++i) { v[i] = in[(size_t)(tid + 256 * i) * HR + j]; s += v[i]; }
  s = blk_sum(s, sm);
  float mu = s * (1.0f / 1024.0f);
  float msj = ms[j];
  float o[4]; float ssq = 0.f;
#pragma unroll
  for (int i = 0; i < 4; ++i) { o[i] = v[i] - msj * mu; ssq += o[i] * o[i]; }
  ssq = blk_sum(ssq, sm);
  float rs = rsqrtf(ssq * (1.0f / 1024.0f) + 1e-5f);
  float g = gamma[j], b = bet[j];
#pragma unroll
  for (int i = 0; i < 4; ++i)
    out[(size_t)(tid + 256 * i) * HR + j] = g * o[i] * rs + b;
}

// In-place row L2 normalize.
__global__ __launch_bounds__(256)
void row_normalize(float* __restrict__ h, int cols) {
  __shared__ float sm[256];
  int c = blockIdx.x, tid = threadIdx.x;
  float* row = h + (size_t)c * cols;
  float s = 0.f;
  for (int j = tid; j < cols; j += 256) { float v = row[j]; s += v * v; }
  s = blk_sum(s, sm);
  float inv = rsqrtf(s);
  for (int j = tid; j < cols; j += 256) row[j] *= inv;
}

// ---------------------------------------------------------------------------
// Host orchestration
// ---------------------------------------------------------------------------
extern "C" void kernel_launch(void* const* d_in, const int* in_sizes, int n_in,
                              void* d_out, int out_size, void* d_ws, size_t ws_size,
                              hipStream_t stream) {
  (void)in_sizes; (void)n_in; (void)out_size; (void)ws_size;

  const float* x       = (const float*)d_in[0];
  const float* q1_w    = (const float*)d_in[1];
  const float* q1_b    = (const float*)d_in[2];
  const float* k1_w    = (const float*)d_in[3];
  const float* k1_b    = (const float*)d_in[4];
  const float* v1_w    = (const float*)d_in[5];
  const float* v1_b    = (const float*)d_in[6];
  const float* e1_w    = (const float*)d_in[7];
  const float* s1_w    = (const float*)d_in[8];
  const float* s1_b    = (const float*)d_in[9];
  const float* gn1_g   = (const float*)d_in[10];
  const float* gn1_b   = (const float*)d_in[11];
  const float* gn1_ms  = (const float*)d_in[12];
  const float* q2_w    = (const float*)d_in[13];
  const float* q2_b    = (const float*)d_in[14];
  const float* k2_w    = (const float*)d_in[15];
  const float* k2_b    = (const float*)d_in[16];
  const float* v2_w    = (const float*)d_in[17];
  const float* v2_b    = (const float*)d_in[18];
  const float* e2_w    = (const float*)d_in[19];
  const float* s2_w    = (const float*)d_in[20];
  const float* s2_b    = (const float*)d_in[21];
  const float* gn2_g   = (const float*)d_in[22];
  const float* gn2_b   = (const float*)d_in[23];
  const float* gn2_ms  = (const float*)d_in[24];

  // -------- workspace carve (bump allocator, 256B aligned) --------
  char* base = (char*)d_ws;
  size_t off = 0;
  auto carve = [&](size_t bytes) -> void* {
    off = (off + 255) & ~(size_t)255;
    void* r = base + off;
    off += bytes;
    return r;
  };

  bf16*  xA   = (bf16*) carve((size_t)kN * 1024 * sizeof(bf16));       // x packed A-frags
  bf16*  h1A  = (bf16*) carve((size_t)kN * 2048 * sizeof(bf16));       // h1 packed A-frags
  float* h1   = (float*)carve((size_t)kN * 2048 * sizeof(float));      // layer-1 output
  float* S    = (float*)carve((size_t)8 * kN * kN * sizeof(float));    // logits / P (8 heads)
  bf16*  PA   = (bf16*) carve((size_t)8 * kN * kN * sizeof(bf16));     // P packed A-frags
  bf16*  VB   = (bf16*) carve((size_t)8 * kN * kCd * sizeof(bf16));    // V packed B-frags
  float* attn = (float*)carve((size_t)kN * 4096 * sizeof(float));      // P@V result
  float* hpre = (float*)carve((size_t)kN * 4096 * sizeof(float));      // pre-GraphNorm
  float* qeB  = (float*)carve((size_t)kN * 8 * sizeof(float));
  float* beta = (float*)carve((size_t)kN * 8 * sizeof(float));
  bf16*  wB   = (bf16*) carve((size_t)4 * 2048 * 4096 * sizeof(bf16)); // packed weights (4 slots)
  float* proj = (float*)carve((size_t)4 * kN * 4096 * sizeof(float));  // q,k,v,skip projections
  bf16*  QA   = (bf16*) carve((size_t)8 * kN * kCd * sizeof(bf16));    // Q packed per head
  bf16*  KB   = (bf16*) carve((size_t)8 * kN * kCd * sizeof(bf16));    // K packed per head

  const size_t wbStride   = (size_t)2048 * 4096;
  const size_t projStride = (size_t)kN * 4096;
  const size_t headAB     = (size_t)kN * kCd;

  auto launch_pack_a = [&](const float* src, int ld, int mTiles, int kTiles, bf16* dst) {
    int total = mTiles * kTiles * 512;
    pack_a_bf16<<<dim3((total + 255) / 256), dim3(256), 0, stream>>>(src, ld, mTiles, kTiles, dst);
  };
  auto launch_pack_b = [&](const float* src, int ld, int kTiles, int nTiles, int trans, bf16* dst) {
    int total = nTiles * kTiles * 512;
    pack_b_bf16<<<dim3((total + 255) / 256), dim3(256), 0, stream>>>(src, ld, kTiles, nTiles, trans, dst);
  };
  auto launch_gemm = [&](const bf16* Ap, const bf16* Bp, const float* bias,
                         float* C, int ldc, int kTiles, int nTiles) {
    // one wave per 2 M-tiles x 4 N-tiles
    gemm_bf16_wmma<<<dim3(kN / 32, nTiles / 4), dim3(32), 0, stream>>>(Ap, Bp, bias, C, ldc, kTiles);
  };

  auto run_layer = [&](const float* inp, bf16* inA, int Kin, int H, int HRout,
                       const float* qw, const float* qb, const float* kw, const float* kb,
                       const float* vw, const float* vb, const float* ew,
                       const float* sw, const float* sb,
                       const float* gg, const float* gb, const float* gms,
                       float* outBuf) {
    const int kT = Kin / 32;
    const int nT = HRout / 16;

    // Pack layer input once.
    launch_pack_a(inp, Kin, kN / 16, kT, inA);

    // q,k,v,skip projections (bias fused in the GEMM epilogue).
    const float* W[4] = {qw, kw, vw, sw};
    const float* B[4] = {qb, kb, vb, sb};
    for (int i = 0; i < 4; ++i) {
      launch_pack_b(W[i], HRout, kT, nT, 0, wB + (size_t)i * wbStride);
      launch_gemm(inA, wB + (size_t)i * wbStride, B[i],
                  proj + (size_t)i * projStride, HRout, kT, nT);
    }
    float* qp = proj + 0 * projStride;
    float* kp = proj + 1 * projStride;
    float* vp = proj + 2 * projStride;
    float* sp = proj + 3 * projStride;

    // qe[c,h] = q . We  (edge logit factor)
    qe_dot<<<dim3(kN * H), dim3(128), 0, stream>>>(qp, ew, qeB, H, HRout);

    // S_h = Q_h @ K_h^T per head
    for (int h = 0; h < H; ++h) {
      launch_pack_a(qp + h * kCd, HRout, kN / 16, kCd / 32, QA + h * headAB);
      launch_pack_b(kp + h * kCd, HRout, kCd / 32, kN / 16, 1, KB + h * headAB);
      launch_gemm(QA + h * headAB, KB + h * headAB, nullptr,
                  S + (size_t)h * kN * kN, kN, kCd / 32, kN / 16);
    }

    // softmax over sources (+ edge term), beta[c,h]
    attn_softmax<<<dim3(H * kN), dim3(256), 0, stream>>>(
        S, x, qeB, beta, H, 1.0f / sqrtf((float)kCd));

    // out_h = P_h @ V_h per head
    for (int h = 0; h < H; ++h) {
      launch_pack_a(S + (size_t)h * kN * kN, kN, kN / 16, kN / 32, PA + (size_t)h * kN * kN);
      launch_pack_b(vp + h * kCd, HRout, kN / 32, kCd / 16, 0, VB + h * headAB);
      launch_gemm(PA + (size_t)h * kN * kN, VB + h * headAB, nullptr,
                  attn + h * kCd, HRout, kN / 32, kCd / 16);
    }

    // + beta*We + skip, then GraphNorm
    int total = kN * HRout;
    attn_epilogue<<<dim3((total + 255) / 256), dim3(256), 0, stream>>>(
        attn, beta, ew, sp, hpre, H, kCd, HRout, total);
    graph_norm<<<dim3(HRout), dim3(256), 0, stream>>>(hpre, outBuf, gg, gb, gms, HRout);
  };

  // Layer 1: in=1024, out=2048, H=4
  run_layer(x, xA, 1024, 4, 2048,
            q1_w, q1_b, k1_w, k1_b, v1_w, v1_b, e1_w, s1_w, s1_b,
            gn1_g, gn1_b, gn1_ms, h1);

  // Layer 2: in=2048, out=4096, H=8 -> GraphNorm straight into d_out
  run_layer(h1, h1A, 2048, 8, 4096,
            q2_w, q2_b, k2_w, k2_b, v2_w, v2_b, e2_w, s2_w, s2_b,
            gn2_g, gn2_b, gn2_ms, (float*)d_out);

  // Final per-node L2 normalization, in place on d_out.
  row_normalize<<<dim3(kN), dim3(256), 0, stream>>>((float*)d_out, 4096);
}
